// MACHAttention_39384850104925
// MI455X (gfx1250) — compile-verified, matrix operand order checked by reading, not run on hardware
//
#include <hip/hip_runtime.h>
#include <hip/hip_bf16.h>

// ---------------- problem constants ----------------
constexpr int Bc = 4, Tc = 4096, Dc = 1024, Hc = 16, DIMc = 64;
constexpr int Lc = 8, KBc = 4, Rc = 16, Sc = 128;

typedef __attribute__((ext_vector_type(16))) __bf16 v16bf;
typedef __attribute__((ext_vector_type(8)))  float  v8f;
typedef __attribute__((ext_vector_type(8)))  unsigned short u16x8;

union BF16Frag { u16x8 h[2]; v16bf v; };

__device__ __forceinline__ unsigned short f2bf(float f) {
  unsigned int u = __float_as_uint(f);
  u += 0x7FFFu + ((u >> 16) & 1u);          // round-to-nearest-even
  return (unsigned short)(u >> 16);
}
__device__ __forceinline__ float bf2f(unsigned short h) {
  return __uint_as_float(((unsigned int)h) << 16);
}

// Load one 16x32 bf16 fragment (A layout; B^T rows load identically) from a
// row-major tile with leading dimension `ld` (elements).
__device__ __forceinline__ v16bf load_frag16x32(const unsigned short* tile, int ld) {
  int lane = threadIdx.x & 31;
  int r  = lane & 15;
  int kb = (lane >> 4) << 3;                 // lanes 0-15: K base 0; 16-31: K base 8
  const unsigned short* p = tile + (size_t)r * ld + kb;
  BF16Frag f;
  f.h[0] = *(const u16x8*)(p);               // K = kb .. kb+7
  f.h[1] = *(const u16x8*)(p + 16);          // K = kb+16 .. kb+23
  return f.v;
}

__device__ __forceinline__ v8f wmma_bf16(v16bf a, v16bf b, v8f c) {
  return __builtin_amdgcn_wmma_f32_16x16x32_bf16(false, a, false, b, (short)0, c,
                                                 false, false);
}

// ---------------- conversion kernels ----------------
__global__ void __launch_bounds__(256) conv_f32_bf16(const float* __restrict__ src,
                                                     unsigned short* __restrict__ dst,
                                                     size_t n) {
  size_t i = (size_t)blockIdx.x * 256 + threadIdx.x;
  if (i < n) dst[i] = f2bf(src[i]);
}

// src [K][N] f32  ->  dst [N][K] bf16  (weight transpose for B fragments)
__global__ void __launch_bounds__(256) conv_transpose_bf16(const float* __restrict__ src,
                                                           unsigned short* __restrict__ dst,
                                                           int K, int N) {
  size_t i = (size_t)blockIdx.x * 256 + threadIdx.x;
  if (i < (size_t)K * N) {
    int n = (int)(i / K), k = (int)(i % K);
    dst[i] = f2bf(src[(size_t)k * N + n]);
  }
}

__global__ void __launch_bounds__(256) zero_f32(float* __restrict__ p, size_t n) {
  size_t i = (size_t)blockIdx.x * 256 + threadIdx.x;
  if (i < n) p[i] = 0.0f;
}

// ---------------- fused QKV GEMM ----------------
// C[M=B*T, N=3072] = xb(bf16) @ Wqkv^T(bf16); epilogue applies mask and scatters
// bf16 results into Q/K/V laid out [b,h,t,d].
// block = 256 threads = 8 waves (2 x 4); block tile 64(M) x 128(N); wave tile 32x32.
__global__ void __launch_bounds__(256) gemm_qkv(const unsigned short* __restrict__ xb,
                                                const unsigned short* __restrict__ wt,
                                                const float* __restrict__ mask,
                                                unsigned short* __restrict__ Qb,
                                                unsigned short* __restrict__ Kb,
                                                unsigned short* __restrict__ Vb) {
  int wave = threadIdx.x >> 5, lane = threadIdx.x & 31;
  int wr = wave >> 2, wc = wave & 3;
  int m0 = blockIdx.y * 64 + wr * 32;
  int n0 = blockIdx.x * 128 + wc * 32;
  v8f acc[2][2] = {};
  for (int k = 0; k < Dc; k += 32) {
    v16bf a0 = load_frag16x32(xb + (size_t)m0 * Dc + k, Dc);
    v16bf a1 = load_frag16x32(xb + (size_t)(m0 + 16) * Dc + k, Dc);
    v16bf b0 = load_frag16x32(wt + (size_t)n0 * Dc + k, Dc);
    v16bf b1 = load_frag16x32(wt + (size_t)(n0 + 16) * Dc + k, Dc);
    acc[0][0] = wmma_bf16(a0, b0, acc[0][0]);
    acc[0][1] = wmma_bf16(a0, b1, acc[0][1]);
    acc[1][0] = wmma_bf16(a1, b0, acc[1][0]);
    acc[1][1] = wmma_bf16(a1, b1, acc[1][1]);
  }
  int colLane = lane & 15, half8 = (lane >> 4) * 8;
#pragma unroll
  for (int i = 0; i < 2; ++i)
#pragma unroll
    for (int j = 0; j < 2; ++j) {
      int n = n0 + 16 * j + colLane;
      int which = n >> 10;
      int hh = (n >> 6) & (Hc - 1);
      int dd = n & (DIMc - 1);
      unsigned short* dst = (which == 0) ? Qb : ((which == 1) ? Kb : Vb);
#pragma unroll
      for (int r = 0; r < 8; ++r) {
        int m = m0 + 16 * i + half8 + r;
        int bb = m >> 12;              // T = 4096
        int tt = m & (Tc - 1);
        float v = acc[i][j][r] * mask[m];
        dst[(((size_t)(bb * Hc + hh)) * Tc + tt) * DIMc + dd] = f2bf(v);
      }
    }
}

// ---------------- bucket assignment + sparse scatter ----------------
// grid (B*H, 4); each block handles 1024 tokens of one (b,h), accumulating
// bucket_K/V and counts into LDS (ds_add_f32), then flushes with global atomics.
__global__ void __launch_bounds__(256) bucket_scatter(const unsigned short* __restrict__ Kb,
                                                      const unsigned short* __restrict__ Vb,
                                                      const float* __restrict__ planesT,
                                                      const float* __restrict__ protosT,
                                                      const float* __restrict__ logit_temp,
                                                      float* __restrict__ gBK,
                                                      float* __restrict__ gBV,
                                                      float* __restrict__ gA) {
  __shared__ float sBK[Sc][DIMc];          // 32 KB
  __shared__ float sBV[Sc][DIMc];          // 32 KB
  __shared__ float sA[Sc];
  __shared__ float sPlanes[DIMc][Lc * KBc];// 8 KB
  __shared__ float sProtos[KBc][Rc];
  __shared__ int   eS[128][16];            // 8 KB
  __shared__ float eW[128][16];            // 8 KB

  int tid = threadIdx.x;
  int bh = blockIdx.x;
  int tBase = blockIdx.y * (Tc / 4);

  for (int i = tid; i < Sc * DIMc; i += 256) {
    sBK[i / DIMc][i % DIMc] = 0.f;
    sBV[i / DIMc][i % DIMc] = 0.f;
  }
  for (int i = tid; i < Sc; i += 256) sA[i] = 0.f;
  for (int i = tid; i < DIMc * Lc * KBc; i += 256) sPlanes[i / 32][i % 32] = planesT[i];
  for (int i = tid; i < KBc * Rc; i += 256) sProtos[i / Rc][i % Rc] = protosT[i];
  __syncthreads();

  float scl = __expf(logit_temp[0]);
  scl = fminf(fmaxf(scl, 0.01f), 20.0f);
  float inv_scl = 1.0f / scl;

  const unsigned short* Krows = Kb + (size_t)bh * Tc * DIMc;
  const unsigned short* Vrows = Vb + (size_t)bh * Tc * DIMc;

  for (int round = 0; round < (Tc / 4) / 128; ++round) {
    int t0 = tBase + round * 128;
    if (tid < 128) {
      int t = t0 + tid;
      const unsigned short* kr = Krows + (size_t)t * DIMc;
      float proj[32];
#pragma unroll
      for (int j = 0; j < 32; ++j) proj[j] = 0.f;
      for (int d = 0; d < DIMc; ++d) {
        float kd = bf2f(kr[d]);
#pragma unroll
        for (int j = 0; j < 32; ++j) proj[j] += kd * sPlanes[d][j];
      }
#pragma unroll
      for (int l = 0; l < Lc; ++l) {
        float th[KBc];
#pragma unroll
        for (int kb = 0; kb < KBc; ++kb) th[kb] = tanhf(proj[l * KBc + kb]) * inv_scl;
        float p[Rc];
        float mx = -3.0e38f;
#pragma unroll
        for (int r = 0; r < Rc; ++r) {
          float s = 0.f;
#pragma unroll
          for (int kb = 0; kb < KBc; ++kb) s += th[kb] * sProtos[kb][r];
          p[r] = s;
          mx = fmaxf(mx, s);
        }
        float Z = 0.f;
#pragma unroll
        for (int r = 0; r < Rc; ++r) { p[r] = __expf(p[r] - mx); Z += p[r]; }
        int i1 = 0; float v1 = p[0];
#pragma unroll
        for (int r = 1; r < Rc; ++r) if (p[r] > v1) { v1 = p[r]; i1 = r; }
        int i2 = -1; float v2 = -3.0e38f;
#pragma unroll
        for (int r = 0; r < Rc; ++r) if (r != i1 && p[r] > v2) { v2 = p[r]; i2 = r; }
        float q1 = v1 / Z, q2 = v2 / Z;
        float den = q1 + q2 + 1e-6f;
        eS[tid][2 * l]     = l * Rc + i1;  eW[tid][2 * l]     = q1 / den;
        eS[tid][2 * l + 1] = l * Rc + i2;  eW[tid][2 * l + 1] = q2 / den;
      }
    }
    __syncthreads();
    // 128 tokens * 16 entries * 64 dims of weighted scatter via LDS atomics
    for (int w = tid; w < 128 * 16 * 64; w += 256) {
      int d  = w & 63;
      int pe = w >> 6;
      int e  = pe & 15;
      int tk = pe >> 4;
      int s = eS[tk][e];
      float wt_ = eW[tk][e];
      size_t row = (size_t)(t0 + tk) * DIMc + d;
      atomicAdd(&sBK[s][d], wt_ * bf2f(Krows[row]));
      atomicAdd(&sBV[s][d], wt_ * bf2f(Vrows[row]));
      if (d == 0) atomicAdd(&sA[s], wt_);
    }
    __syncthreads();
  }
  for (int i = tid; i < Sc * DIMc; i += 256) {
    atomicAdd(&gBK[(size_t)bh * Sc * DIMc + i], sBK[i / DIMc][i % DIMc]);
    atomicAdd(&gBV[(size_t)bh * Sc * DIMc + i], sBV[i / DIMc][i % DIMc]);
  }
  for (int i = tid; i < Sc; i += 256) atomicAdd(&gA[bh * Sc + i], sA[i]);
}

// ---------------- bucket normalize -> bf16 (K: [s][d], V: transposed [d][s]) --
__global__ void __launch_bounds__(256) bucket_norm(const float* __restrict__ gBK,
                                                   const float* __restrict__ gBV,
                                                   const float* __restrict__ gA,
                                                   unsigned short* __restrict__ bKb,
                                                   unsigned short* __restrict__ bVtb) {
  size_t i = (size_t)blockIdx.x * 256 + threadIdx.x;
  if (i >= (size_t)Bc * Hc * Sc * DIMc) return;
  int d = (int)(i & (DIMc - 1));
  size_t sd = i >> 6;
  int s  = (int)(sd & (Sc - 1));
  int bh = (int)(sd >> 7);
  float inv = 1.0f / (gA[bh * Sc + s] + 1e-6f);
  bKb[i] = f2bf(gBK[i] * inv);
  bVtb[((size_t)bh * DIMc + d) * Sc + s] = f2bf(gBV[i] * inv);
}

// ---------------- bucketed attention (two WMMA GEMMs + softmax) ----------------
// grid (B*H, T/128); block 256 = 8 waves; each wave owns a 16-row Q tile.
__global__ void __launch_bounds__(256) attn_kernel(const unsigned short* __restrict__ Qb,
                                                   const unsigned short* __restrict__ bKb,
                                                   const unsigned short* __restrict__ bVtb,
                                                   unsigned short* __restrict__ attnb) {
  __shared__ unsigned short ldsP[8][16][Sc];   // 32 KB
  int wave = threadIdx.x >> 5, lane = threadIdx.x & 31;
  int bh = blockIdx.x;
  int b = bh >> 4, h = bh & (Hc - 1);
  int t0 = blockIdx.y * 128 + wave * 16;
  const unsigned short* Qtile = Qb + ((size_t)bh * Tc + t0) * DIMc;
  const unsigned short* bKp = bKb + (size_t)bh * Sc * DIMc;   // [s][d]
  const unsigned short* bVp = bVtb + (size_t)bh * DIMc * Sc;  // [d][s]

  v8f sc[8] = {};
#pragma unroll
  for (int kk = 0; kk < DIMc; kk += 32) {
    v16bf a = load_frag16x32(Qtile + kk, DIMc);
#pragma unroll
    for (int j = 0; j < 8; ++j) {
      v16bf bf = load_frag16x32(bKp + (size_t)(j * 16) * DIMc + kk, DIMc);
      sc[j] = wmma_bf16(a, bf, sc[j]);
    }
  }
  int colLane = lane & 15, half8 = (lane >> 4) * 8;
#pragma unroll
  for (int r = 0; r < 8; ++r) {
    float mx = -3.0e38f;
#pragma unroll
    for (int j = 0; j < 8; ++j) mx = fmaxf(mx, sc[j][r]);
    mx = fmaxf(mx, __shfl_xor(mx, 1));
    mx = fmaxf(mx, __shfl_xor(mx, 2));
    mx = fmaxf(mx, __shfl_xor(mx, 4));
    mx = fmaxf(mx, __shfl_xor(mx, 8));
    mx *= 0.125f;                              // 1/sqrt(DIM)
    float e[8], sum = 0.f;
#pragma unroll
    for (int j = 0; j < 8; ++j) { e[j] = __expf(sc[j][r] * 0.125f - mx); sum += e[j]; }
    sum += __shfl_xor(sum, 1);
    sum += __shfl_xor(sum, 2);
    sum += __shfl_xor(sum, 4);
    sum += __shfl_xor(sum, 8);
    float inv = 1.0f / sum;
#pragma unroll
    for (int j = 0; j < 8; ++j)
      ldsP[wave][half8 + r][j * 16 + colLane] = f2bf(e[j] * inv);
  }
  __syncthreads();

  v8f o[4] = {};
#pragma unroll
  for (int kk = 0; kk < Sc; kk += 32) {
    v16bf a = load_frag16x32(&ldsP[wave][0][0] + kk, Sc);
#pragma unroll
    for (int j = 0; j < 4; ++j) {
      v16bf bf = load_frag16x32(bVp + (size_t)(j * 16) * Sc + kk, Sc);
      o[j] = wmma_bf16(a, bf, o[j]);
    }
  }
#pragma unroll
  for (int j = 0; j < 4; ++j) {
    int d = j * 16 + colLane;
#pragma unroll
    for (int r = 0; r < 8; ++r) {
      int t = t0 + half8 + r;
      attnb[((size_t)(b * Tc + t)) * Dc + h * DIMc + d] = f2bf(o[j][r]);
    }
  }
}

// ---------------- output GEMM with bias ----------------
__global__ void __launch_bounds__(256) gemm_out(const unsigned short* __restrict__ ab,
                                                const unsigned short* __restrict__ wt,
                                                const float* __restrict__ bias,
                                                float* __restrict__ out) {
  int wave = threadIdx.x >> 5, lane = threadIdx.x & 31;
  int wr = wave >> 2, wc = wave & 3;
  int m0 = blockIdx.y * 64 + wr * 32;
  int n0 = blockIdx.x * 128 + wc * 32;
  v8f acc[2][2] = {};
  for (int k = 0; k < Dc; k += 32) {
    v16bf a0 = load_frag16x32(ab + (size_t)m0 * Dc + k, Dc);
    v16bf a1 = load_frag16x32(ab + (size_t)(m0 + 16) * Dc + k, Dc);
    v16bf b0 = load_frag16x32(wt + (size_t)n0 * Dc + k, Dc);
    v16bf b1 = load_frag16x32(wt + (size_t)(n0 + 16) * Dc + k, Dc);
    acc[0][0] = wmma_bf16(a0, b0, acc[0][0]);
    acc[0][1] = wmma_bf16(a0, b1, acc[0][1]);
    acc[1][0] = wmma_bf16(a1, b0, acc[1][0]);
    acc[1][1] = wmma_bf16(a1, b1, acc[1][1]);
  }
  int colLane = lane & 15, half8 = (lane >> 4) * 8;
#pragma unroll
  for (int i = 0; i < 2; ++i)
#pragma unroll
    for (int j = 0; j < 2; ++j) {
      int n = n0 + 16 * j + colLane;
      float bv = bias[n];
#pragma unroll
      for (int r = 0; r < 8; ++r) {
        int m = m0 + 16 * i + half8 + r;
        out[(size_t)m * Dc + n] = acc[i][j][r] + bv;
      }
    }
}

// ---------------- workspace layout (bytes) ----------------
constexpr size_t NBT = (size_t)Bc * Tc;                       // 16384 rows
constexpr size_t OFF_XB    = 0;                               // bf16 x  [M,K]
constexpr size_t OFF_WQKVT = OFF_XB    + NBT * Dc * 2;        // bf16 [3N,K]
constexpr size_t OFF_WOUTT = OFF_WQKVT + (size_t)3 * Dc * Dc * 2;
constexpr size_t OFF_QB    = OFF_WOUTT + (size_t)Dc * Dc * 2; // bf16 [b,h,t,d]
constexpr size_t OFF_KB    = OFF_QB    + NBT * Dc * 2;
constexpr size_t OFF_VB    = OFF_KB    + NBT * Dc * 2;
constexpr size_t OFF_GBK   = OFF_VB    + NBT * Dc * 2;        // f32 accumulators
constexpr size_t OFF_GBV   = OFF_GBK   + (size_t)Bc * Hc * Sc * DIMc * 4;
constexpr size_t OFF_GA    = OFF_GBV   + (size_t)Bc * Hc * Sc * DIMc * 4;
constexpr size_t OFF_BKB   = OFF_GA    + (size_t)Bc * Hc * Sc * 4;
constexpr size_t OFF_BVTB  = OFF_BKB   + (size_t)Bc * Hc * Sc * DIMc * 2;
constexpr size_t OFF_ATTNB = OFF_BVTB  + (size_t)Bc * Hc * Sc * DIMc * 2;

extern "C" void kernel_launch(void* const* d_in, const int* in_sizes, int n_in,
                              void* d_out, int out_size, void* d_ws, size_t ws_size,
                              hipStream_t stream) {
  const float* x       = (const float*)d_in[0];
  const float* mask    = (const float*)d_in[1];
  const float* Wq      = (const float*)d_in[2];
  const float* Wk      = (const float*)d_in[3];
  const float* Wv      = (const float*)d_in[4];
  const float* Wout    = (const float*)d_in[5];
  const float* b_out   = (const float*)d_in[6];
  const float* planesT = (const float*)d_in[7];
  const float* protosT = (const float*)d_in[8];
  const float* ltemp   = (const float*)d_in[9];
  float* out = (float*)d_out;

  char* ws = (char*)d_ws;
  unsigned short* xb    = (unsigned short*)(ws + OFF_XB);
  unsigned short* wqkvt = (unsigned short*)(ws + OFF_WQKVT);
  unsigned short* woutt = (unsigned short*)(ws + OFF_WOUTT);
  unsigned short* Qb    = (unsigned short*)(ws + OFF_QB);
  unsigned short* Kb    = (unsigned short*)(ws + OFF_KB);
  unsigned short* Vb    = (unsigned short*)(ws + OFF_VB);
  float* gBK            = (float*)(ws + OFF_GBK);
  float* gBV            = (float*)(ws + OFF_GBV);
  float* gA             = (float*)(ws + OFF_GA);
  unsigned short* bKb   = (unsigned short*)(ws + OFF_BKB);
  unsigned short* bVtb  = (unsigned short*)(ws + OFF_BVTB);
  unsigned short* attnb = (unsigned short*)(ws + OFF_ATTNB);

  size_t nX = NBT * Dc;
  conv_f32_bf16<<<dim3((unsigned)(nX / 256)), 256, 0, stream>>>(x, xb, nX);
  unsigned gW = (unsigned)((size_t)Dc * Dc / 256);
  conv_transpose_bf16<<<gW, 256, 0, stream>>>(Wq, wqkvt, Dc, Dc);
  conv_transpose_bf16<<<gW, 256, 0, stream>>>(Wk, wqkvt + (size_t)Dc * Dc, Dc, Dc);
  conv_transpose_bf16<<<gW, 256, 0, stream>>>(Wv, wqkvt + (size_t)2 * Dc * Dc, Dc, Dc);
  conv_transpose_bf16<<<gW, 256, 0, stream>>>(Wout, woutt, Dc, Dc);

  gemm_qkv<<<dim3(3 * Dc / 128, (unsigned)(NBT / 64)), 256, 0, stream>>>(
      xb, wqkvt, mask, Qb, Kb, Vb);

  size_t nZero = (size_t)2 * Bc * Hc * Sc * DIMc + (size_t)Bc * Hc * Sc;
  zero_f32<<<dim3((unsigned)((nZero + 255) / 256)), 256, 0, stream>>>(gBK, nZero);

  bucket_scatter<<<dim3(Bc * Hc, 4), 256, 0, stream>>>(
      Kb, Vb, planesT, protosT, ltemp, gBK, gBV, gA);

  bucket_norm<<<dim3((unsigned)((size_t)Bc * Hc * Sc * DIMc / 256)), 256, 0, stream>>>(
      gBK, gBV, gA, bKb, bVtb);

  attn_kernel<<<dim3(Bc * Hc, Tc / 128), 256, 0, stream>>>(Qb, bKb, bVtb, attnb);

  gemm_out<<<dim3(Dc / 128, (unsigned)(NBT / 64)), 256, 0, stream>>>(
      attnb, woutt, b_out, out);
}
